// _NonLocalBlockND_26645977105104
// MI455X (gfx1250) — matrix operand order repaired and checked by reading, not compiled, not run on hardware
//
#include <hip/hip_runtime.h>
#include <hip/hip_bf16.h>

typedef __attribute__((ext_vector_type(16))) __bf16 v16bf;
typedef __attribute__((ext_vector_type(8)))  float  v8f;

#define B_    4
#define C_    256
#define CI_   128
#define H_    96
#define W_    96
#define NPIX  (H_*W_)        // 9216
#define MPOOL (48*48)        // 2304
#define QT_PER_B (NPIX/16)   // 576 query tiles per batch image

__device__ __forceinline__ v8f wmma_bf16(v16bf a, v16bf b, v8f c) {
  // D = A(16x32 bf16) * B(32x16 bf16) + C(16x16 f32)
  return __builtin_amdgcn_wmma_f32_16x16x32_bf16(false, a, false, b, (short)0, c, false, false);
}
// A-fragment K index for element e, half h (contiguous: k = 16h + e, matches B fills)
__device__ __forceinline__ int kA(int e, int h) { return e + (h << 4); }

// ---------------------------------------------------------------------------
// Kernel 0: one-time fp32 -> bf16 weight conversion (weights are L2-resident;
// doing this once removes per-wave v_cvt_pk_bf16 work from all GEMM kernels).
// ---------------------------------------------------------------------------
__global__ void cvt_kernel(const float* __restrict__ src, __bf16* __restrict__ dst, int n) {
  const int i = blockIdx.x * 256 + threadIdx.x;
  if (i < n) dst[i] = (__bf16)src[i];
}

// ---------------------------------------------------------------------------
// Kernel 1: projection. For a 16-pixel tile, compute theta/g/phi 1x1 convs
// (K=256) with bf16 WMMA, sharing the x B-fragments across all three weights.
// ---------------------------------------------------------------------------
__global__ void proj_kernel(const float* __restrict__ x,
                            const __bf16* __restrict__ Wt, const float* __restrict__ bt,
                            const __bf16* __restrict__ Wg, const float* __restrict__ bg,
                            const __bf16* __restrict__ Wp, const float* __restrict__ bp,
                            __bf16* __restrict__ theta,
                            float* __restrict__ gfull, float* __restrict__ pfull) {
  const int lane = threadIdx.x & 31, wave = threadIdx.x >> 5;
  const int t  = blockIdx.x * 8 + wave;          // 2304 tiles total
  const int b  = t / QT_PER_B;
  const int p0 = (t % QT_PER_B) * 16;
  const int col = lane & 15, h = lane >> 4;
  const float* xb = x + (size_t)b * C_ * NPIX;

  // B fragments of x: B[c, p] = x[c, p0+col]; coalesced across lanes
  v16bf bx[8];
#pragma unroll
  for (int kt = 0; kt < 8; ++kt)
#pragma unroll
    for (int e = 0; e < 16; ++e) {
      int c = e + 16 * h + 32 * kt;
      bx[kt][e] = (__bf16)xb[(size_t)c * NPIX + p0 + col];
    }

  const size_t prow = (size_t)(b * NPIX + p0 + col) * CI_;
#pragma unroll
  for (int cot = 0; cot < 8; ++cot) {
    const int co0 = cot * 16;
    v8f at = {}, ag = {}, ap = {};
#pragma unroll
    for (int kt = 0; kt < 8; ++kt) {
      v16bf a;
      const __bf16* wr = Wt + (size_t)(co0 + col) * C_ + 32 * kt;
#pragma unroll
      for (int e = 0; e < 16; ++e) a[e] = wr[kA(e, h)];
      at = wmma_bf16(a, bx[kt], at);
      wr = Wg + (size_t)(co0 + col) * C_ + 32 * kt;
#pragma unroll
      for (int e = 0; e < 16; ++e) a[e] = wr[kA(e, h)];
      ag = wmma_bf16(a, bx[kt], ag);
      wr = Wp + (size_t)(co0 + col) * C_ + 32 * kt;
#pragma unroll
      for (int e = 0; e < 16; ++e) a[e] = wr[kA(e, h)];
      ap = wmma_bf16(a, bx[kt], ap);
    }
#pragma unroll
    for (int r = 0; r < 8; ++r) {
      const int co = co0 + r + 8 * h;
      theta[prow + co] = (__bf16)(at[r] + bt[co]);
      gfull[prow + co] = ag[r] + bg[co];
      pfull[prow + co] = ap[r] + bp[co];
    }
  }
}

// ---------------------------------------------------------------------------
// Kernel 2: 2x2 maxpool of g/phi conv outputs; emit phi as (M,128) bf16 and
// g transposed as (128,M) bf16 (the layouts the flash kernel's frags want).
// ---------------------------------------------------------------------------
__global__ void pool_kernel(const float* __restrict__ gfull, const float* __restrict__ pfull,
                            __bf16* __restrict__ gT, __bf16* __restrict__ phi) {
  const int idx = blockIdx.x * 256 + threadIdx.x;
  if (idx >= B_ * MPOOL * CI_) return;
  const int c = idx % CI_;
  const int m = (idx / CI_) % MPOOL;
  const int b = idx / (CI_ * MPOOL);
  const int h2 = m / 48, w2 = m % 48;
  const int p00 = (2 * h2) * W_ + 2 * w2;
  const size_t base = (size_t)b * NPIX * CI_;
  const size_t i0 = base + (size_t)p00 * CI_ + c;
  float gm = fmaxf(fmaxf(gfull[i0], gfull[i0 + CI_]),
                   fmaxf(gfull[i0 + (size_t)W_ * CI_], gfull[i0 + (size_t)(W_ + 1) * CI_]));
  float pm = fmaxf(fmaxf(pfull[i0], pfull[i0 + CI_]),
                   fmaxf(pfull[i0 + (size_t)W_ * CI_], pfull[i0 + (size_t)(W_ + 1) * CI_]));
  gT[(size_t)b * CI_ * MPOOL + (size_t)c * MPOOL + m] = (__bf16)gm;
  phi[(size_t)(b * MPOOL + m) * CI_ + c] = (__bf16)pm;
}

// ---------------------------------------------------------------------------
// Kernel 3: flash attention. One wave owns 16 queries; streams keys 32 at a
// time; online softmax with per-lane stats (q = lane%16) + one xor-16 shuffle.
// Score tile is produced as D[m,q] then re-laid to a B-fragment (m=K, q=N)
// with half-swap shuffles, so P @ g runs straight back through WMMA.
// Output y written directly in bf16 (only consumer is the bf16 WMMA in zconv).
// ---------------------------------------------------------------------------
__global__ void flash_kernel(const __bf16* __restrict__ theta, const __bf16* __restrict__ phi,
                             const __bf16* __restrict__ gT, __bf16* __restrict__ y) {
  const int lane = threadIdx.x & 31, wave = threadIdx.x >> 5;
  const int t  = blockIdx.x * 8 + wave;           // 2304 query tiles
  const int b  = t / QT_PER_B;
  const int q0 = (t % QT_PER_B) * 16;
  const int col = lane & 15, h = lane >> 4;
  const __bf16* th = theta + (size_t)b * NPIX * CI_;
  const __bf16* ph = phi + (size_t)b * MPOOL * CI_;
  const __bf16* g  = gT  + (size_t)b * CI_ * MPOOL;

  // theta as B fragments (K=c, N=q); reused for every key tile
  v16bf bq[4];
#pragma unroll
  for (int kt = 0; kt < 4; ++kt)
#pragma unroll
    for (int e = 0; e < 16; ++e)
      bq[kt][e] = th[(size_t)(q0 + col) * CI_ + e + 16 * h + 32 * kt];

  float mstat = -1e30f, lsum = 0.f;
  v8f acc[8];
#pragma unroll
  for (int i = 0; i < 8; ++i) acc[i] = (v8f){};

  for (int m0 = 0; m0 < MPOOL; m0 += 32) {
    // scores S[m, q] for 32 keys, K = 128 channels
    v8f S0 = {}, S1 = {};
#pragma unroll
    for (int kt = 0; kt < 4; ++kt) {
      v16bf a0, a1;
      const __bf16* pr0 = ph + (size_t)(m0 + col) * CI_ + 32 * kt;
      const __bf16* pr1 = ph + (size_t)(m0 + 16 + col) * CI_ + 32 * kt;
#pragma unroll
      for (int e = 0; e < 16; ++e) { a0[e] = pr0[kA(e, h)]; a1[e] = pr1[kA(e, h)]; }
      S0 = wmma_bf16(a0, bq[kt], S0);
      S1 = wmma_bf16(a1, bq[kt], S1);
    }
    // online softmax: stats per q (= lane&15); combine halves with xor-16
    float tmax = -1e30f;
#pragma unroll
    for (int r = 0; r < 8; ++r) tmax = fmaxf(tmax, fmaxf(S0[r], S1[r]));
    tmax = fmaxf(tmax, __shfl_xor(tmax, 16));
    const float mnew = fmaxf(mstat, tmax);
    const float corr = __expf(mstat - mnew);
    mstat = mnew;
    lsum *= corr;
#pragma unroll
    for (int i = 0; i < 8; ++i)
#pragma unroll
      for (int r = 0; r < 8; ++r) acc[i][r] *= corr;
#pragma unroll
    for (int r = 0; r < 8; ++r) {
      S0[r] = __expf(S0[r] - mnew);
      S1[r] = __expf(S1[r] - mnew);
      lsum += S0[r] + S1[r];
    }
    // D-layout P -> B-fragment (K=m 0..31, N=q): swap halves via shuffles
    v16bf pb;
#pragma unroll
    for (int e = 0; e < 8; ++e) {
      const float o0 = __shfl_xor(S0[e], 16);
      const float o1 = __shfl_xor(S1[e], 16);
      pb[e]     = (__bf16)(h == 0 ? S0[e] : o1);
      pb[e + 8] = (__bf16)(h == 0 ? o0 : S1[e]);
    }
    // Y[c, q] += gT[c, m] * P[m, q]
#pragma unroll
    for (int ct = 0; ct < 8; ++ct) {
      v16bf ag;
      const __bf16* gr = g + (size_t)(ct * 16 + col) * MPOOL + m0;
#pragma unroll
      for (int e = 0; e < 16; ++e) ag[e] = gr[kA(e, h)];
      acc[ct] = wmma_bf16(ag, pb, acc[ct]);
    }
  }
  lsum += __shfl_xor(lsum, 16);
  const float inv = 1.f / lsum;
  const size_t qrow = (size_t)(b * NPIX + q0 + col) * CI_;
#pragma unroll
  for (int ct = 0; ct < 8; ++ct)
#pragma unroll
    for (int r = 0; r < 8; ++r)
      y[qrow + ct * 16 + r + 8 * h] = (__bf16)(acc[ct][r] * inv);
}

// ---------------------------------------------------------------------------
// Kernel 4: output 1x1 conv (128 -> 256), WMMA over K=128, z in NCHW layout.
// ---------------------------------------------------------------------------
__global__ void zconv_kernel(const __bf16* __restrict__ y, const __bf16* __restrict__ Ww,
                             const float* __restrict__ bw, float* __restrict__ z) {
  const int lane = threadIdx.x & 31, wave = threadIdx.x >> 5;
  const int t  = blockIdx.x * 8 + wave;
  const int b  = t / QT_PER_B;
  const int p0 = (t % QT_PER_B) * 16;
  const int col = lane & 15, h = lane >> 4;

  v16bf by[4];
#pragma unroll
  for (int kt = 0; kt < 4; ++kt)
#pragma unroll
    for (int e = 0; e < 16; ++e)
      by[kt][e] = y[(size_t)(b * NPIX + p0 + col) * CI_ + e + 16 * h + 32 * kt];

#pragma unroll
  for (int cot = 0; cot < 16; ++cot) {
    v8f a = {};
#pragma unroll
    for (int kt = 0; kt < 4; ++kt) {
      v16bf aw;
      const __bf16* wr = Ww + (size_t)(cot * 16 + col) * CI_ + 32 * kt;
#pragma unroll
      for (int e = 0; e < 16; ++e) aw[e] = wr[kA(e, h)];
      a = wmma_bf16(aw, by[kt], a);
    }
#pragma unroll
    for (int r = 0; r < 8; ++r) {
      const int co = cot * 16 + r + 8 * h;
      z[(size_t)b * C_ * NPIX + (size_t)co * NPIX + p0 + col] = a[r] + bw[co];
    }
  }
}

// ---------------------------------------------------------------------------
// Kernel 5: deterministic per-channel BN batch statistics (one block/channel).
// ---------------------------------------------------------------------------
__global__ void stats_kernel(const float* __restrict__ z, float* __restrict__ stats) {
  const int co = blockIdx.x;
  __shared__ float ssum[256], ssq[256];
  float s = 0.f, sq = 0.f;
  for (int i = threadIdx.x; i < B_ * NPIX; i += 256) {
    const int b = i / NPIX, p = i % NPIX;
    const float v = z[(size_t)b * C_ * NPIX + (size_t)co * NPIX + p];
    s += v; sq += v * v;
  }
  ssum[threadIdx.x] = s; ssq[threadIdx.x] = sq;
  __syncthreads();
  for (int off = 128; off > 0; off >>= 1) {
    if (threadIdx.x < off) {
      ssum[threadIdx.x] += ssum[threadIdx.x + off];
      ssq[threadIdx.x]  += ssq[threadIdx.x + off];
    }
    __syncthreads();
  }
  if (threadIdx.x == 0) {
    const float n = (float)(B_ * NPIX);
    const float mean = ssum[0] / n;
    stats[co * 2]     = mean;
    stats[co * 2 + 1] = ssq[0] / n - mean * mean;
  }
}

// ---------------------------------------------------------------------------
// Kernel 6: BN apply + residual.
// ---------------------------------------------------------------------------
__global__ void final_kernel(const float* __restrict__ z, const float* __restrict__ x,
                             const float* __restrict__ stats, const float* __restrict__ gamma,
                             const float* __restrict__ beta, float* __restrict__ out) {
  const size_t idx = (size_t)blockIdx.x * 256 + threadIdx.x;
  const int co = (int)((idx / NPIX) % C_);
  const float mean = stats[co * 2], var = stats[co * 2 + 1];
  out[idx] = (z[idx] - mean) * rsqrtf(var + 1e-5f) * gamma[co] + beta[co] + x[idx];
}

extern "C" void kernel_launch(void* const* d_in, const int* in_sizes, int n_in,
                              void* d_out, int out_size, void* d_ws, size_t ws_size,
                              hipStream_t stream) {
  const float* x     = (const float*)d_in[0];
  const float* Wg    = (const float*)d_in[1];
  const float* bg    = (const float*)d_in[2];
  const float* Wt    = (const float*)d_in[3];
  const float* bt    = (const float*)d_in[4];
  const float* Wp    = (const float*)d_in[5];
  const float* bp    = (const float*)d_in[6];
  const float* Ww    = (const float*)d_in[7];
  const float* bw    = (const float*)d_in[8];
  const float* gamma = (const float*)d_in[9];
  const float* beta  = (const float*)d_in[10];
  float* out = (float*)d_out;

  char* ws = (char*)d_ws;
  __bf16* theta = (__bf16*)(ws + 0);                      //  9,437,184 B
  float*  gfull = (float*)(ws + 9437184);                 // 18,874,368 B
  float*  pfull = (float*)(ws + 9437184 + 18874368);      // 18,874,368 B
  __bf16* gT    = (__bf16*)(ws + 47185920);               //  2,359,296 B
  __bf16* phi   = (__bf16*)(ws + 49545216);               //  2,359,296 B
  __bf16* ybf   = (__bf16*)(ws + 51904512);               //  9,437,184 B
  float*  z     = (float*)(ws + 9437184);                 // reuses gfull+pfull (dead by then)
  float*  stats = (float*)(ws + 61341696);                //      2,048 B
  __bf16* Wtb   = (__bf16*)(ws + 61343744);               //     65,536 B
  __bf16* Wgb   = (__bf16*)(ws + 61409280);               //     65,536 B
  __bf16* Wpb   = (__bf16*)(ws + 61474816);               //     65,536 B
  __bf16* Wwb   = (__bf16*)(ws + 61540352);               //     65,536 B

  const int nw = CI_ * C_;  // 32768 elements per projection weight (and Ww)
  cvt_kernel<<<(nw + 255) / 256, 256, 0, stream>>>(Wt, Wtb, nw);
  cvt_kernel<<<(nw + 255) / 256, 256, 0, stream>>>(Wg, Wgb, nw);
  cvt_kernel<<<(nw + 255) / 256, 256, 0, stream>>>(Wp, Wpb, nw);
  cvt_kernel<<<(nw + 255) / 256, 256, 0, stream>>>(Ww, Wwb, nw);

  proj_kernel<<<288, 256, 0, stream>>>(x, Wtb, bt, Wgb, bg, Wpb, bp, theta, gfull, pfull);
  pool_kernel<<<(B_ * MPOOL * CI_ + 255) / 256, 256, 0, stream>>>(gfull, pfull, gT, phi);
  flash_kernel<<<288, 256, 0, stream>>>(theta, phi, gT, ybf);
  zconv_kernel<<<288, 256, 0, stream>>>(ybf, Wwb, bw, z);
  stats_kernel<<<256, 256, 0, stream>>>(z, stats);
  final_kernel<<<(B_ * C_ * NPIX + 255) / 256, 256, 0, stream>>>(z, x, stats, gamma, beta, out);
}
